// SeqV2_1992864825930
// MI455X (gfx1250) — compile-verified, matrix operand order checked by reading, not run on hardware
//
#include <hip/hip_runtime.h>
#include <cstdint>

// ---------------------------------------------------------------------------
// Types for CDNA5 WMMA (wave32, 16x16x32 bf16 -> f32 accum)
// ---------------------------------------------------------------------------
typedef __attribute__((ext_vector_type(16))) __bf16 v16bf;
typedef __attribute__((ext_vector_type(8)))  __bf16 v8bf;
typedef __attribute__((ext_vector_type(8)))  float  v8f;

#define N_VARS 128
#define BATCH  256
#define ZDIM   512
#define HID    1024
#define GATES  4096   // 4*HID
#define VAL    256
#define M_ALL  (N_VARS * BATCH)   // 32768

// ---------------------------------------------------------------------------
// Helpers
// ---------------------------------------------------------------------------
__device__ __forceinline__ unsigned short f2bf(float f) {
    unsigned int u = __float_as_uint(f);
    return (unsigned short)((u + 0x7FFFu + ((u >> 16) & 1u)) >> 16);  // RNE
}

__device__ __forceinline__ float sigm(float x) { return 1.0f / (1.0f + __expf(-x)); }
__device__ __forceinline__ float tanh_f(float x) { return 2.0f / (1.0f + __expf(-2.0f * x)) - 1.0f; }

// Load one 16-bit WMMA A/B fragment (16 rows x 32 K, K-contiguous rows).
// Lane r<16 holds K {0..7,16..23}; lane r+16 holds K {8..15,24..31}.
// Works for both global and LDS pointers (addrspace inferred after inlining).
__device__ __forceinline__ v16bf frag16(const unsigned short* p) {
    v8bf lo = *reinterpret_cast<const v8bf*>(p);
    v8bf hi = *reinterpret_cast<const v8bf*>(p + 16);
    return __builtin_shufflevector(lo, hi, 0,1,2,3,4,5,6,7,8,9,10,11,12,13,14,15);
}

#define WMMA_BF16(a, b, c) \
    __builtin_amdgcn_wmma_f32_16x16x32_bf16(false, (a), false, (b), (short)0, (c), false, false)

// ---------------------------------------------------------------------------
// Elementwise kernels
// ---------------------------------------------------------------------------
__global__ void k_f32_to_bf16(const float* __restrict__ in,
                              unsigned short* __restrict__ out, long long n) {
    long long i = (long long)blockIdx.x * blockDim.x + threadIdx.x;
    long long s = (long long)gridDim.x * blockDim.x;
    for (; i < n; i += s) out[i] = f2bf(in[i]);
}

__global__ void k_zero32(unsigned int* __restrict__ p, long long n) {
    long long i = (long long)blockIdx.x * blockDim.x + threadIdx.x;
    long long s = (long long)gridDim.x * blockDim.x;
    for (; i < n; i += s) p[i] = 0u;
}

// ---------------------------------------------------------------------------
// Kernel: xg = z @ W_ih^T + b_ih + b_hh   (M=32768, K=512, N=4096)
// Wave computes 16x64 (4 tiles); K-loop software-pipelined (prefetch k+32).
// ---------------------------------------------------------------------------
__global__ void k_gemm_xg(const unsigned short* __restrict__ A,   // [32768,512] bf16
                          const unsigned short* __restrict__ B,   // [4096,512]  bf16
                          const float* __restrict__ b_ih,
                          const float* __restrict__ b_hh,
                          float* __restrict__ xg) {               // [32768,4096]
    const int lane = threadIdx.x & 31;
    const int w    = threadIdx.x >> 5;
    const int r    = lane & 15;
    const int hi   = lane >> 4;
    const int m0   = blockIdx.y * 64 + w * 16;
    const int n0   = blockIdx.x * 64;

    const unsigned short* arow = A + (size_t)(m0 + r) * ZDIM + hi * 8;
    const unsigned short* brow = B + (size_t)(n0 + r) * ZDIM + hi * 8;

    v8f acc[4];
    #pragma unroll
    for (int i = 0; i < 4; ++i) acc[i] = v8f{};

    v16bf a  = frag16(arow);
    v16bf b0 = frag16(brow);
    v16bf b1 = frag16(brow + 16 * ZDIM);
    v16bf b2 = frag16(brow + 32 * ZDIM);
    v16bf b3 = frag16(brow + 48 * ZDIM);

    #pragma unroll 2
    for (int k = 0; k < ZDIM; k += 32) {
        const int kn = (k + 32) & (ZDIM - 1);   // wraps to 0 on last iter
        v16bf an  = frag16(arow + kn);
        v16bf b0n = frag16(brow + kn);
        v16bf b1n = frag16(brow + 16 * ZDIM + kn);
        v16bf b2n = frag16(brow + 32 * ZDIM + kn);
        v16bf b3n = frag16(brow + 48 * ZDIM + kn);
        acc[0] = WMMA_BF16(a, b0, acc[0]);
        acc[1] = WMMA_BF16(a, b1, acc[1]);
        acc[2] = WMMA_BF16(a, b2, acc[2]);
        acc[3] = WMMA_BF16(a, b3, acc[3]);
        a = an; b0 = b0n; b1 = b1n; b2 = b2n; b3 = b3n;
    }

    const int rowoff = hi * 8;
    #pragma unroll
    for (int i = 0; i < 4; ++i) {
        const int col  = n0 + 16 * i + r;
        const float bias = b_ih[col] + b_hh[col];
        #pragma unroll
        for (int e = 0; e < 8; ++e)
            xg[(size_t)(m0 + rowoff + e) * GATES + col] = acc[i][e] + bias;
    }
}

// ---------------------------------------------------------------------------
// Kernel: one LSTM step. gates = xg[t] + h @ W_hh^T, fused activations.
// Block stages its 128 KB W_hh slice (16 hidden cols x 4 gates x 1024 K) in
// LDS once; all 4 waves feed B fragments from ds_load_b128 (bank-padded).
// Wave computes 32 batch rows x 16 hidden cols x 4 gates = 8 WMMAs/K-step.
// grid = (HID/16=64, BATCH/128=2), block = 128 (4 waves).
// ---------------------------------------------------------------------------
#define SROW (HID + 8)   // row stride in elements; 2064 B = 516 dwords = 4 mod 64 banks

__global__ void k_lstm_step(const unsigned short* __restrict__ Whh,   // [4096,1024] bf16
                            const float* __restrict__ xg_t,           // [256,4096]
                            const unsigned short* __restrict__ hprev, // [256,1024] bf16
                            unsigned short* __restrict__ hnext,       // [256,1024] bf16
                            float* __restrict__ cst) {                // [256,1024]
    __shared__ unsigned short sw[64 * SROW];     // 64 rows = 4 gates x 16 cols, ~129 KB

    const int lane = threadIdx.x & 31;
    const int w    = threadIdx.x >> 5;
    const int r    = lane & 15;
    const int hi   = lane >> 4;
    const int m0   = blockIdx.y * 128 + w * 32;  // batch rows (32 per wave)
    const int j0   = blockIdx.x * 16;            // hidden cols

    // --- cooperative W_hh slice -> LDS (64 rows x 1024 bf16, uint4 chunks) ---
    for (int i = threadIdx.x; i < 64 * (HID / 8); i += 128) {
        const int row = i >> 7;                  // 0..63  (i / 128 chunks per row)
        const int c   = i & 127;                 // uint4 index within row
        const int g   = row >> 4;
        const int rr  = row & 15;
        const uint4 v = *reinterpret_cast<const uint4*>(
            Whh + (size_t)(g * HID + j0 + rr) * HID + c * 8);
        *reinterpret_cast<uint4*>(&sw[row * SROW + c * 8]) = v;
    }
    __syncthreads();

    const unsigned short* arow0 = hprev + (size_t)(m0 + r) * HID + hi * 8;
    const unsigned short* arow1 = arow0 + 16 * HID;
    const unsigned short* sb0 = &sw[(0 * 16 + r) * SROW + hi * 8];
    const unsigned short* sb1 = &sw[(1 * 16 + r) * SROW + hi * 8];
    const unsigned short* sb2 = &sw[(2 * 16 + r) * SROW + hi * 8];
    const unsigned short* sb3 = &sw[(3 * 16 + r) * SROW + hi * 8];

    v8f acc[2][4];
    #pragma unroll
    for (int p = 0; p < 2; ++p)
        #pragma unroll
        for (int g = 0; g < 4; ++g) acc[p][g] = v8f{};

    v16bf a0 = frag16(arow0);
    v16bf a1 = frag16(arow1);
    v16bf b0 = frag16(sb0);
    v16bf b1 = frag16(sb1);
    v16bf b2 = frag16(sb2);
    v16bf b3 = frag16(sb3);

    #pragma unroll 2
    for (int k = 0; k < HID; k += 32) {
        const int kn = (k + 32) & (HID - 1);     // wraps to 0 on last iter
        v16bf a0n = frag16(arow0 + kn);
        v16bf a1n = frag16(arow1 + kn);
        v16bf b0n = frag16(sb0 + kn);
        v16bf b1n = frag16(sb1 + kn);
        v16bf b2n = frag16(sb2 + kn);
        v16bf b3n = frag16(sb3 + kn);
        acc[0][0] = WMMA_BF16(a0, b0, acc[0][0]);
        acc[0][1] = WMMA_BF16(a0, b1, acc[0][1]);
        acc[0][2] = WMMA_BF16(a0, b2, acc[0][2]);
        acc[0][3] = WMMA_BF16(a0, b3, acc[0][3]);
        acc[1][0] = WMMA_BF16(a1, b0, acc[1][0]);
        acc[1][1] = WMMA_BF16(a1, b1, acc[1][1]);
        acc[1][2] = WMMA_BF16(a1, b2, acc[1][2]);
        acc[1][3] = WMMA_BF16(a1, b3, acc[1][3]);
        a0 = a0n; a1 = a1n; b0 = b0n; b1 = b1n; b2 = b2n; b3 = b3n;
    }

    const int j = j0 + r;
    const int rowoff = hi * 8;
    #pragma unroll
    for (int p = 0; p < 2; ++p) {
        #pragma unroll
        for (int e = 0; e < 8; ++e) {
            const int bb = m0 + p * 16 + rowoff + e;
            const float* xr = xg_t + (size_t)bb * GATES;
            const float gi = sigm(acc[p][0][e]   + xr[j]);
            const float gf = sigm(acc[p][1][e]   + xr[HID + j]);
            const float gg = tanh_f(acc[p][2][e] + xr[2 * HID + j]);
            const float go = sigm(acc[p][3][e]   + xr[3 * HID + j]);
            const size_t ci = (size_t)bb * HID + j;
            const float cn = gf * cst[ci] + gi * gg;
            cst[ci] = cn;
            hnext[ci] = f2bf(go * tanh_f(cn));
        }
    }
}

// ---------------------------------------------------------------------------
// Kernel: out = |H_all @ W_lin^T + b_lin|   (M=32768, K=1024, N=256)
// ---------------------------------------------------------------------------
__global__ void k_gemm_out(const unsigned short* __restrict__ H,   // [32768,1024] bf16
                           const unsigned short* __restrict__ B,   // [256,1024]   bf16
                           const float* __restrict__ b_lin,
                           float* __restrict__ out) {              // [32768,256]
    const int lane = threadIdx.x & 31;
    const int w    = threadIdx.x >> 5;
    const int r    = lane & 15;
    const int hi   = lane >> 4;
    const int m0   = blockIdx.y * 64 + w * 16;
    const int n0   = blockIdx.x * 64;

    const unsigned short* arow = H + (size_t)(m0 + r) * HID + hi * 8;
    const unsigned short* brow = B + (size_t)(n0 + r) * HID + hi * 8;

    v8f acc[4];
    #pragma unroll
    for (int i = 0; i < 4; ++i) acc[i] = v8f{};

    v16bf a  = frag16(arow);
    v16bf b0 = frag16(brow);
    v16bf b1 = frag16(brow + 16 * HID);
    v16bf b2 = frag16(brow + 32 * HID);
    v16bf b3 = frag16(brow + 48 * HID);

    #pragma unroll 2
    for (int k = 0; k < HID; k += 32) {
        const int kn = (k + 32) & (HID - 1);
        v16bf an  = frag16(arow + kn);
        v16bf b0n = frag16(brow + kn);
        v16bf b1n = frag16(brow + 16 * HID + kn);
        v16bf b2n = frag16(brow + 32 * HID + kn);
        v16bf b3n = frag16(brow + 48 * HID + kn);
        acc[0] = WMMA_BF16(a, b0, acc[0]);
        acc[1] = WMMA_BF16(a, b1, acc[1]);
        acc[2] = WMMA_BF16(a, b2, acc[2]);
        acc[3] = WMMA_BF16(a, b3, acc[3]);
        a = an; b0 = b0n; b1 = b1n; b2 = b2n; b3 = b3n;
    }

    const int rowoff = hi * 8;
    #pragma unroll
    for (int i = 0; i < 4; ++i) {
        const int col = n0 + 16 * i + r;
        const float bias = b_lin[col];
        #pragma unroll
        for (int e = 0; e < 8; ++e)
            out[(size_t)(m0 + rowoff + e) * VAL + col] = fabsf(acc[i][e] + bias);
    }
}

// ---------------------------------------------------------------------------
// Host launcher
// ---------------------------------------------------------------------------
extern "C" void kernel_launch(void* const* d_in, const int* in_sizes, int n_in,
                              void* d_out, int out_size, void* d_ws, size_t ws_size,
                              hipStream_t stream) {
    (void)in_sizes; (void)n_in; (void)out_size; (void)ws_size;

    const float* z     = (const float*)d_in[0];
    const float* W_ih  = (const float*)d_in[1];
    const float* W_hh  = (const float*)d_in[2];
    const float* b_ih  = (const float*)d_in[3];
    const float* b_hh  = (const float*)d_in[4];
    const float* W_lin = (const float*)d_in[5];
    const float* b_lin = (const float*)d_in[6];

    uint8_t* ws = (uint8_t*)d_ws;
    size_t off = 0;
    float* xg = (float*)(ws + off);                  off += (size_t)M_ALL * GATES * 4;      // 512 MB
    unsigned short* zbf   = (unsigned short*)(ws + off); off += (size_t)M_ALL * ZDIM * 2;   // 32 MB
    unsigned short* wihb  = (unsigned short*)(ws + off); off += (size_t)GATES * ZDIM * 2;   // 4 MB
    unsigned short* whhb  = (unsigned short*)(ws + off); off += (size_t)GATES * HID * 2;    // 8 MB
    unsigned short* wlinb = (unsigned short*)(ws + off); off += (size_t)VAL * HID * 2;      // 0.5 MB
    unsigned short* hbuf  = (unsigned short*)(ws + off); off += (size_t)(N_VARS + 1) * BATCH * HID * 2; // 64.5 MB
    float* cst = (float*)(ws + off);                 off += (size_t)BATCH * HID * 4;        // 1 MB

    const size_t HSLOT = (size_t)BATCH * HID;        // elements per h slot

    // 1) bf16 conversions of all GEMM operands
    k_f32_to_bf16<<<4096, 256, 0, stream>>>(z,     zbf,   (long long)M_ALL * ZDIM);
    k_f32_to_bf16<<<1024, 256, 0, stream>>>(W_ih,  wihb,  (long long)GATES * ZDIM);
    k_f32_to_bf16<<<2048, 256, 0, stream>>>(W_hh,  whhb,  (long long)GATES * HID);
    k_f32_to_bf16<<<128,  256, 0, stream>>>(W_lin, wlinb, (long long)VAL * HID);

    // 2) zero c and h0 (hbuf slot 0)
    k_zero32<<<256, 256, 0, stream>>>((unsigned int*)cst,  (long long)HSLOT);
    k_zero32<<<128, 256, 0, stream>>>((unsigned int*)hbuf, (long long)(HSLOT / 2));

    // 3) input projections for all steps (big parallel GEMM)
    k_gemm_xg<<<dim3(GATES / 64, M_ALL / 64), 128, 0, stream>>>(zbf, wihb, b_ih, b_hh, xg);

    // 4) sequential recurrence: 128 fused gate-GEMM + activation steps
    for (int t = 0; t < N_VARS; ++t) {
        k_lstm_step<<<dim3(HID / 16, BATCH / 128), 128, 0, stream>>>(
            whhb,
            xg + (size_t)t * BATCH * GATES,
            hbuf + (size_t)t * HSLOT,
            hbuf + (size_t)(t + 1) * HSLOT,
            cst);
    }

    // 5) output projection for all steps at once
    k_gemm_out<<<dim3(VAL / 64, M_ALL / 64), 128, 0, stream>>>(
        hbuf + HSLOT, wlinb, b_lin, (float*)d_out);
}
